// VersorLinear_23502061043883
// MI455X (gfx1250) — matrix operand order repaired and checked by reading, not compiled, not run on hardware
//
#include <hip/hip_runtime.h>
#include <math.h>

// ---- problem dims ----
#define BDIM_ 2
#define SDIM_ 1024
#define INDIM_ 256
#define OUTDIM_ 256
#define M_ (BDIM_ * SDIM_)   // 2048
#define K_ (INDIM_ * 32)     // 8192
#define N_ (OUTDIM_ * 32)    // 8192

// ---- GEMM tiling ----
#define BM 128
#define BN 256
#define BK 32
#define LDSK (BK + 8)              // 40 ushorts = 80B padded row (TDM pad 4dw/16dw)
#define A_WORDS (BM * LDSK)        // 5120 ushorts (10240 B)
#define B_WORDS (BN * LDSK)        // 10240 ushorts (20480 B)
#define BUF_WORDS (A_WORDS + B_WORDS)

typedef __attribute__((ext_vector_type(16))) __bf16 v16bf;
typedef __attribute__((ext_vector_type(8)))  __bf16 v8bf;
typedef __attribute__((ext_vector_type(8)))  float  v8f;
typedef __attribute__((ext_vector_type(4)))  unsigned int v4u;
typedef __attribute__((ext_vector_type(8)))  int v8i;
typedef __attribute__((ext_vector_type(4)))  int v4i;

static __device__ __forceinline__ unsigned short f32_to_bf16_rne(float f) {
    unsigned int u = __float_as_uint(f);
    return (unsigned short)((u + 0x7FFFu + ((u >> 16) & 1u)) >> 16);
}

// ---------------- kernel 1: x fp32 -> bf16 ----------------
__global__ void cvt_x_bf16(const float* __restrict__ x,
                           unsigned short* __restrict__ xb, int n) {
    int i = blockIdx.x * blockDim.x + threadIdx.x;
    if (i < n) xb[i] = f32_to_bf16_rne(x[i]);
}

// ---------------- kernel 2: build W' (N-major: [N][K]) in bf16 ----------------
// W_bt[n*K + kk] = sign(j=l^k, l) * weight[o][i][l^k],  n=o*32+k, kk=i*32+l
__global__ void build_wbt(const float* __restrict__ w,
                          unsigned short* __restrict__ wbt) {
    int e  = blockIdx.x * blockDim.x + threadIdx.x;
    int kk = e & (K_ - 1);
    int n  = e >> 13;                                  // K_ = 2^13
    int l  = kk & 31;
    int i  = kk >> 5;
    int k  = n & 31;
    int o  = n >> 5;
    int j  = l ^ k;
    int swaps = 0;
    int t = j >> 1;
    while (t) { swaps += __popc(t & l); t >>= 1; }
    int neg = (swaps & 1) ^ (((j & l) >> 4) & 1);      // metric: e5 (bit4) -> -1
    float val = w[(o * INDIM_ + i) * 32 + j];
    val = neg ? -val : val;
    wbt[e] = f32_to_bf16_rne(val);
}

// ---------------- TDM: 2-D tile (rows x 64B) global -> LDS, padded rows ----------------
// D# per CDNA5 ISA: group0 {count,lds,global,type}, group1
// {data_size=4B, pad 4dw every 16dw, tensor_dim0=4096dw, tile_dim0=16dw, tile_dim1=rows,
//  tensor_dim0_stride=4096dw}
static __device__ __forceinline__ void tdm_load_2d(unsigned lds_off,
                                                   const void* gptr,
                                                   int rows) {
    unsigned long long ga = (unsigned long long)(uintptr_t)gptr;
    v4u g0;
    g0.x = 1u;                                              // count=1, user mode
    g0.y = lds_off;                                         // LDS byte address
    g0.z = (unsigned)(ga & 0xffffffffu);                    // global_addr[31:0]
    g0.w = (unsigned)((ga >> 32) & 0x01ffffffu) | (2u << 30); // addr[56:32] | type=2
    v8i g1;
    g1[0] = (int)((2u << 16) | (1u << 20) | (3u << 22) | (3u << 25));
    //            data_size=4B  pad_en      every 16dw     pad 4dw
    g1[1] = (int)(4096u << 16);      // tensor_dim0 = 4096 dwords (lo16 in [31:16])
    g1[2] = (int)(0x8000u << 16);    // tensor_dim0 hi=0 ; tensor_dim1 lo16 = 32768
    g1[3] = (int)(16u << 16);        // tensor_dim1 hi=0 ; tile_dim0 = 16 dwords
    g1[4] = rows;                    // tile_dim1 = rows ; tile_dim2 = 0
    g1[5] = 4096;                    // tensor_dim0_stride = 4096 dwords
    g1[6] = 0;
    g1[7] = 0;
    v4i z4 = {0, 0, 0, 0};
    v8i z8 = {0, 0, 0, 0, 0, 0, 0, 0};
    __builtin_amdgcn_tensor_load_to_lds(g0, g1, z4, z4, z8, 0);
}

// ---------------- kernel 3: bf16 WMMA GEMM, TDM-staged, double-buffered ----------------
__global__ __launch_bounds__(256) void gemm_bf16(
    const unsigned short* __restrict__ A,   // [M][K] bf16
    const unsigned short* __restrict__ Bt,  // [N][K] bf16 (B^T layout)
    float* __restrict__ C) {                // [M][N] f32
    __shared__ unsigned short smem[2][BUF_WORDS];           // 2 x 30720 B

    const int tid  = threadIdx.x;
    const int lane = tid & 31;
    const int wave = tid >> 5;   // 0..7
    const int wm   = wave & 1;   // 2 slabs of 64 rows
    const int wn   = wave >> 1;  // 4 slabs of 64 cols
    const int m0   = blockIdx.y * BM;
    const int n0   = blockIdx.x * BN;

    const int h = lane >> 4;     // lane half (ISA fragment layout)
    const int r = lane & 15;     // row/col within 16

    const unsigned ldsA0 = (unsigned)(uintptr_t)(const void*)&smem[0][0];
    const unsigned ldsB0 = (unsigned)(uintptr_t)(const void*)&smem[0][A_WORDS];
    const unsigned bufBytes = BUF_WORDS * 2u;

    v8f c[4][4] = {};

    // prologue: DMA first tiles into buffer 0
    if (wave == 0)      tdm_load_2d(ldsA0, A  + (size_t)m0 * K_, BM);
    else if (wave == 1) tdm_load_2d(ldsB0, Bt + (size_t)n0 * K_, BN);
    __builtin_amdgcn_s_wait_tensorcnt(0);
    __syncthreads();

    int p = 0;
    for (int kb = 0; kb < K_; kb += BK, p ^= 1) {
        // issue next tile DMA into the other buffer (overlaps with compute)
        if (kb + BK < K_) {
            if (wave == 0)
                tdm_load_2d(ldsA0 + (unsigned)(p ^ 1) * bufBytes,
                            A + (size_t)m0 * K_ + kb + BK, BM);
            else if (wave == 1)
                tdm_load_2d(ldsB0 + (unsigned)(p ^ 1) * bufBytes,
                            Bt + (size_t)n0 * K_ + kb + BK, BN);
        }

        const unsigned short* Abuf = &smem[p][0];
        const unsigned short* Bbuf = &smem[p][A_WORDS];

        union Frag { v16bf v; v8bf h8[2]; };
        Frag a[4], b[4];
        // A fragment (16x32 bf16): lane half h holds K {8h..8h+7, 16+8h..16+8h+7}
#pragma unroll
        for (int mi = 0; mi < 4; ++mi) {
            const unsigned short* base = Abuf + (wm * 64 + mi * 16 + r) * LDSK;
            a[mi].h8[0] = *(const v8bf*)(base + 8 * h);
            a[mi].h8[1] = *(const v8bf*)(base + 16 + 8 * h);
        }
        // B fragment (32x16 bf16, col-major source): lane half h holds K {8h..8h+7, 16+8h..16+8h+7}
#pragma unroll
        for (int ni = 0; ni < 4; ++ni) {
            const unsigned short* base = Bbuf + (wn * 64 + ni * 16 + r) * LDSK;
            b[ni].h8[0] = *(const v8bf*)(base + 8 * h);
            b[ni].h8[1] = *(const v8bf*)(base + 16 + 8 * h);
        }
#pragma unroll
        for (int mi = 0; mi < 4; ++mi)
#pragma unroll
            for (int ni = 0; ni < 4; ++ni)
                c[mi][ni] = __builtin_amdgcn_wmma_f32_16x16x32_bf16(
                    false, a[mi].v, false, b[ni].v,
                    (short)0, c[mi][ni], false, false);

        __builtin_amdgcn_s_wait_tensorcnt(0);  // next buffer landed (waves 0/1)
        __syncthreads();                       // publish to all waves
    }

    // Store: D layout — VGPR rr, lane: (M = rr + 8*(lane>>4), N = lane&15)
#pragma unroll
    for (int mi = 0; mi < 4; ++mi)
#pragma unroll
        for (int ni = 0; ni < 4; ++ni) {
            int col = n0 + wn * 64 + ni * 16 + r;
#pragma unroll
            for (int rr = 0; rr < 8; ++rr) {
                int row = m0 + wm * 64 + mi * 16 + rr + 8 * h;
                C[(size_t)row * N_ + col] = c[mi][ni][rr];
            }
        }
}

// ---------------- kernel 4: per-multivector normalization (in place) ----------------
__global__ void normalize_mv(float* __restrict__ out, int ngroups) {
    int g = blockIdx.x * blockDim.x + threadIdx.x;
    if (g >= ngroups) return;
    float4* p = (float4*)(out + (size_t)g * 32);
    float4 v[8];
    float s = 0.f;
#pragma unroll
    for (int i = 0; i < 8; ++i) {
        v[i] = p[i];
        s += v[i].x * v[i].x + v[i].y * v[i].y + v[i].z * v[i].z + v[i].w * v[i].w;
    }
    float inv = 1.0f / sqrtf(s + 1e-6f);
#pragma unroll
    for (int i = 0; i < 8; ++i) {
        v[i].x *= inv; v[i].y *= inv; v[i].z *= inv; v[i].w *= inv;
        p[i] = v[i];
    }
}

extern "C" void kernel_launch(void* const* d_in, const int* in_sizes, int n_in,
                              void* d_out, int out_size, void* d_ws, size_t ws_size,
                              hipStream_t stream) {
    const float* x = (const float*)d_in[0];   // [2,1024,256,32] f32
    const float* w = (const float*)d_in[1];   // [256,256,32] f32
    float* out = (float*)d_out;               // [2,1024,256,32] f32

    unsigned short* xb  = (unsigned short*)d_ws;                       // 32 MiB
    unsigned short* wbt = (unsigned short*)((char*)d_ws +
                                            (size_t)M_ * K_ * sizeof(unsigned short)); // 128 MiB

    cvt_x_bf16<<<(M_ * K_) / 256, 256, 0, stream>>>(x, xb, M_ * K_);
    build_wbt<<<(int)(((size_t)N_ * K_) / 256), 256, 0, stream>>>(w, wbt);

    dim3 grid(N_ / BN, M_ / BM);  // 32 x 16
    gemm_bf16<<<grid, 256, 0, stream>>>(xb, wbt, out);

    normalize_mv<<<(M_ * OUTDIM_) / 256, 256, 0, stream>>>(out, M_ * OUTDIM_);
}
